// LAAttention_8744553415147
// MI455X (gfx1250) — matrix operand order repaired and checked by reading, not compiled, not run on hardware
//
#include <hip/hip_runtime.h>

// ---------------------------------------------------------------------------
// LAAttention for MI455X (gfx1250, wave32, WMMA 16x16x32 bf16 / f32 accum)
//
// Shapes: B=2, S=2048, D=2048, H=32, HD=64.  M = B*S = 4096.
// Pipeline:
//   1) f32 -> bf16 converts (X, Wq, Wk, Wv, Wo)
//   2) GEMM + fused epilogue:  Q = norm(elu((X Wq^T + bq)*0.125)+1)   [bf16, BHSD]
//                              K = norm(elu( X Wk^T + bk       )+1)   [bf16, BHSD]
//                              V =           X Wv^T + bv              [bf16, BHSD]
//   3) causal tiled attention: O = (tril(QK^T)*0.125) V               [bf16, BSD]
//   4) GEMM:  out = O Wo^T + bo                                       [f32]
//
// CDNA5-specific paths: v_wmma_f32_16x16x32_bf16 compute,
// global_load_async_to_lds_b128 staging (ASYNCcnt) with LDS double-buffering.
// ---------------------------------------------------------------------------

typedef __attribute__((ext_vector_type(16))) __bf16 v16bf;
typedef __attribute__((ext_vector_type(8)))  float  v8f;

union Frag {
    v16bf v;
    unsigned short h[16];
    unsigned int   u[8];
};

__device__ __forceinline__ unsigned short f2bf(float f) {
    unsigned int u = __float_as_uint(f);
    u += 0x7FFFu + ((u >> 16) & 1u);           // round-to-nearest-even
    return (unsigned short)(u >> 16);
}

__device__ __forceinline__ v8f wmma_bf16(v16bf a, v16bf b, v8f c) {
    // D = A(16x32) * B(32x16) + C(16x16 f32)
    return __builtin_amdgcn_wmma_f32_16x16x32_bf16(
        false, a, false, b, (short)0, c, false, false);
}

// 32-bit LDS address (generic pointer low bits = LDS offset)
__device__ __forceinline__ unsigned int lds_addr32(const void* p) {
    return (unsigned int)(unsigned long long)p;
}

// Direct global -> LDS async copy, 16B per lane. GVS mode: mem = saddr + vaddr.
// Tracked by ASYNCcnt (CDNA5 ISA).
__device__ __forceinline__ void async_copy_b128(unsigned int lds_off,
                                                const void* sbase,
                                                unsigned int byte_off) {
    asm volatile("global_load_async_to_lds_b128 %0, %1, %2"
                 :
                 : "v"(lds_off), "v"(byte_off), "s"(sbase)
                 : "memory");
}

__device__ __forceinline__ void wait_async0() {
    asm volatile("s_wait_asynccnt 0x0" ::: "memory");
}

// A-fragment (16x32 bf16, MxK). lane<16: K = 0..15; lane>=16: K = 16..31
template<int STRIDE>
__device__ __forceinline__ v16bf load_a_frag(const unsigned short* lds,
                                             int row, int kbase, int lane) {
    const int half = (lane >> 4) & 1;
    const unsigned short* p = lds + row * STRIDE + kbase;
    const int o0 = half * 16;
    const int o1 = half * 16 + 8;
    Frag f;
#pragma unroll
    for (int j = 0; j < 4; ++j) {
        f.u[j]     = *(const unsigned int*)(p + o0 + 2 * j);
        f.u[4 + j] = *(const unsigned int*)(p + o1 + 2 * j);
    }
    return f.v;
}

// B-fragment (32x16 bf16, KxN). lane holds N=lane&15 fixed; lanes 0-15: K 0..15,
// lanes 16-31: K 16..31. LDS layout is [n][k] (k contiguous).
template<int STRIDE>
__device__ __forceinline__ v16bf load_b_frag(const unsigned short* lds,
                                             int nrow, int kbase, int lane) {
    const int half = (lane >> 4) & 1;
    const unsigned short* p = lds + nrow * STRIDE + kbase + half * 16;
    Frag f;
#pragma unroll
    for (int j = 0; j < 8; ++j)
        f.u[j] = *(const unsigned int*)(p + 2 * j);
    return f.v;
}

// ---------------------------------------------------------------------------
// f32 -> bf16 convert
// ---------------------------------------------------------------------------
__global__ void f32_to_bf16(const float* __restrict__ in,
                            unsigned short* __restrict__ out, int n) {
    int i = blockIdx.x * blockDim.x + threadIdx.x;
    const int stride = gridDim.x * blockDim.x;
    for (; i < n; i += stride) out[i] = f2bf(in[i]);
}

// ---------------------------------------------------------------------------
// GEMM  C[M,N] = A[M,K](bf16) * W[N,K]^T(bf16)  + fused epilogue
//   MODE 0: Q feature map  (bias, *0.125, elu+1, row-normalize over head, bf16 BHSD)
//   MODE 1: K feature map  (bias, elu+1, row-normalize over head, bf16 BHSD)
//   MODE 2: V bias         (bias, bf16 BHSD)
//   MODE 3: out projection (bias, f32 [M,N])
// Block: 256 threads (8 waves). Tile: 128(M) x 64(N), K-step 32.
// Double-buffered LDS fed by global_load_async_to_lds_b128; one barrier/step.
// ---------------------------------------------------------------------------
#define LSTRIDE 40   // ushorts per LDS row: 80B -> 16B aligned, bank-spread

template<int MODE>
__global__ __launch_bounds__(256)
void gemm_bf16_ep(const unsigned short* __restrict__ A,
                  const unsigned short* __restrict__ Wb,
                  const float* __restrict__ bias,
                  void* __restrict__ outp,
                  int M, int N, int K) {
    __shared__ unsigned short Al[2][128 * LSTRIDE];
    __shared__ unsigned short Bl[2][64 * LSTRIDE];

    const int tid  = threadIdx.x;
    const int lane = tid & 31;
    const int wave = tid >> 5;            // 0..7
    const int half = (lane >> 4) & 1;
    const int ln15 = lane & 15;
    const int nBase = blockIdx.x * 64;
    const int mBase = blockIdx.y * 128;

    // async-stage one K-step tile pair into buffer `buf`
    auto stage = [&](int buf, int kb) {
        // A tile 128x32 (2 x b128 per thread)
#pragma unroll
        for (int i = 0; i < 2; ++i) {
            int c = tid + i * 256;                 // 0..511
            int row = c >> 2, c8 = c & 3;
            async_copy_b128(lds_addr32(&Al[buf][row * LSTRIDE + c8 * 8]),
                            A,
                            (unsigned)(((size_t)(mBase + row) * K + kb + c8 * 8) * 2));
        }
        // W tile 64x32 (native [N,K] layout -> B-operand [n][k])
        {
            int row = tid >> 2, c8 = tid & 3;
            async_copy_b128(lds_addr32(&Bl[buf][row * LSTRIDE + c8 * 8]),
                            Wb,
                            (unsigned)(((size_t)(nBase + row) * K + kb + c8 * 8) * 2));
        }
    };

    v8f acc[4];
#pragma unroll
    for (int i = 0; i < 4; ++i)
#pragma unroll
        for (int j = 0; j < 8; ++j) acc[i][j] = 0.0f;

    stage(0, 0);
    wait_async0();
    __syncthreads();

    int cur = 0;
    for (int kb = 0; kb < K; kb += 32) {
        const int nxt = cur ^ 1;
        if (kb + 32 < K) stage(nxt, kb + 32);      // overlap copy with compute

        const unsigned short* Ac = Al[cur];
        const unsigned short* Bc = Bl[cur];
        v16bf af = load_a_frag<LSTRIDE>(Ac, wave * 16 + ln15, 0, lane);
#pragma unroll
        for (int nt = 0; nt < 4; ++nt) {
            v16bf bf = load_b_frag<LSTRIDE>(Bc, nt * 16 + ln15, 0, lane);
            acc[nt] = wmma_bf16(af, bf, acc[nt]);
        }

        wait_async0();                             // next tile landed (this wave)
        __syncthreads();                           // all waves: reads done, tile ready
        cur = nxt;
    }

    // ---- epilogue ----
    float bi[4];
#pragma unroll
    for (int nt = 0; nt < 4; ++nt) bi[nt] = bias[nBase + nt * 16 + ln15];

    if (MODE == 0 || MODE == 1) {
        // feature map + per-head row normalization; N-tile == one head (HD=64)
        unsigned short* o = (unsigned short*)outp;
        float vals[4][8];
#pragma unroll
        for (int nt = 0; nt < 4; ++nt)
#pragma unroll
            for (int r = 0; r < 8; ++r) {
                float x = acc[nt][r] + bi[nt];
                if (MODE == 0) x *= 0.125f;                 // HD^-0.5 pre-scaling
                vals[nt][r] = (x > 0.0f) ? (x + 1.0f) : __expf(x);  // elu(x)+1
            }
        const int h = nBase >> 6;                            // head index
#pragma unroll
        for (int r = 0; r < 8; ++r) {
            float s = vals[0][r] + vals[1][r] + vals[2][r] + vals[3][r];
            s += __shfl_xor(s, 1);
            s += __shfl_xor(s, 2);
            s += __shfl_xor(s, 4);
            s += __shfl_xor(s, 8);                           // 16-lane half reduce
            const float inv = 1.0f / (s + 1e-4f);
            const int m = mBase + wave * 16 + r + half * 8;  // global row in [0,4096)
            const int b = m >> 11;                           // S = 2048
            const int srow = m & 2047;
            const size_t base = ((size_t)(b * 32 + h) * 2048 + srow) * 64;
#pragma unroll
            for (int nt = 0; nt < 4; ++nt)
                o[base + nt * 16 + ln15] = f2bf(vals[nt][r] * inv);
        }
    } else if (MODE == 2) {
        unsigned short* o = (unsigned short*)outp;
        const int h = nBase >> 6;
#pragma unroll
        for (int r = 0; r < 8; ++r) {
            const int m = mBase + wave * 16 + r + half * 8;
            const int b = m >> 11;
            const int srow = m & 2047;
            const size_t base = ((size_t)(b * 32 + h) * 2048 + srow) * 64;
#pragma unroll
            for (int nt = 0; nt < 4; ++nt)
                o[base + nt * 16 + ln15] = f2bf(acc[nt][r] + bi[nt]);
        }
    } else {
        float* o = (float*)outp;
#pragma unroll
        for (int r = 0; r < 8; ++r) {
            const int m = mBase + wave * 16 + r + half * 8;
#pragma unroll
            for (int nt = 0; nt < 4; ++nt)
                o[(size_t)m * N + nBase + nt * 16 + ln15] = acc[nt][r] + bi[nt];
        }
    }
}

// ---------------------------------------------------------------------------
// Causal attention:  O = (tril(Q K^T) * 0.125) V    per (b,h)
// Grid: (S/128, B*H). Block 256 threads (8 waves); each wave owns 16 s-rows.
// T-blocks of 64, only lower-triangular blocks visited. Q/K tiles staged via
// async-to-LDS; V tile transposed through VGPRs. attn tile bounced through
// LDS as bf16 to re-feed the P*V WMMA.
// ---------------------------------------------------------------------------
#define QSTR 72      // 144B rows: 16B aligned, bank-spread

__global__ __launch_bounds__(256)
void la_attention(const unsigned short* __restrict__ Qf,
                  const unsigned short* __restrict__ Kf,
                  const unsigned short* __restrict__ Vf,
                  unsigned short* __restrict__ O) {
    __shared__ unsigned short Qatt[128 * QSTR];  // Q staging, then attn tile
    __shared__ unsigned short Kl[64 * QSTR];     // [t][d]
    __shared__ unsigned short Vl[64 * QSTR];     // transposed [d][t]

    const int tid  = threadIdx.x;
    const int lane = tid & 31;
    const int wave = tid >> 5;
    const int half = (lane >> 4) & 1;
    const int ln15 = lane & 15;
    const int sBlk = blockIdx.x;
    const int bh   = blockIdx.y;                 // b*32 + h
    const int sBase = sBlk * 128;
    const size_t headBase = (size_t)bh * 2048 * 64;
    const unsigned short* Qh = Qf + headBase;
    const unsigned short* Kh = Kf + headBase;
    const unsigned short* Vh = Vf + headBase;

    // stage Q tile 128x64 (async) and keep as 2 A-fragments (d = 0..31, 32..63)
#pragma unroll
    for (int i = 0; i < 4; ++i) {
        int c = tid + i * 256;                   // 0..1023
        int row = c >> 3, c8 = c & 7;
        async_copy_b128(lds_addr32(&Qatt[row * QSTR + c8 * 8]),
                        Qh, (unsigned)(((sBase + row) * 64 + c8 * 8) * 2));
    }
    wait_async0();
    __syncthreads();
    v16bf aq0 = load_a_frag<QSTR>(Qatt, wave * 16 + ln15, 0,  lane);
    v16bf aq1 = load_a_frag<QSTR>(Qatt, wave * 16 + ln15, 32, lane);

    v8f oacc[4];
#pragma unroll
    for (int i = 0; i < 4; ++i)
#pragma unroll
        for (int j = 0; j < 8; ++j) oacc[i][j] = 0.0f;

    const int tbMax = 2 * sBlk + 1;              // causal: only blocks <= diagonal
    for (int tb = 0; tb <= tbMax; ++tb) {
        const int tBase = tb * 64;
        __syncthreads();                         // retire prior reads of Kl/Vl/Qatt
        // stage K tile 64x64 ([t][d], async direct copy)
#pragma unroll
        for (int i = 0; i < 2; ++i) {
            int c = tid + i * 256;
            int row = c >> 3, c8 = c & 7;
            async_copy_b128(lds_addr32(&Kl[row * QSTR + c8 * 8]),
                            Kh, (unsigned)(((tBase + row) * 64 + c8 * 8) * 2));
        }
        // stage V tile transposed -> Vl[d][t] (VGPR transpose)
#pragma unroll
        for (int i = 0; i < 2; ++i) {
            int c = tid + i * 256;
            int row = c >> 3, c8 = c & 7;        // row = t, cols = d chunk
            union { uint4 q; unsigned short us[8]; } u;
            u.q = *(const uint4*)(Vh + (size_t)(tBase + row) * 64 + c8 * 8);
#pragma unroll
            for (int j = 0; j < 8; ++j)
                Vl[(c8 * 8 + j) * QSTR + row] = u.us[j];
        }
        wait_async0();
        __syncthreads();

        // attn = Q K^T  (K dim = d, two 32-wide steps)
        v8f pacc[4];
#pragma unroll
        for (int i = 0; i < 4; ++i)
#pragma unroll
            for (int j = 0; j < 8; ++j) pacc[i][j] = 0.0f;
#pragma unroll
        for (int nt = 0; nt < 4; ++nt) {
            v16bf b0 = load_b_frag<QSTR>(Kl, nt * 16 + ln15, 0,  lane);
            pacc[nt] = wmma_bf16(aq0, b0, pacc[nt]);
            v16bf b1 = load_b_frag<QSTR>(Kl, nt * 16 + ln15, 32, lane);
            pacc[nt] = wmma_bf16(aq1, b1, pacc[nt]);
        }

        // scale * multiplicative causal mask, bf16 into Qatt (attn tile)
        const int sRow0 = sBase + wave * 16;
#pragma unroll
        for (int nt = 0; nt < 4; ++nt)
#pragma unroll
            for (int r = 0; r < 8; ++r) {
                const int sg = sRow0 + r + half * 8;
                const int tg = tBase + nt * 16 + ln15;
                const float v = (tg <= sg) ? pacc[nt][r] * 0.125f : 0.0f;
                Qatt[(wave * 16 + r + half * 8) * QSTR + nt * 16 + ln15] = f2bf(v);
            }
        __syncthreads();

        // O += attn * V   (K dim = t, two 32-wide steps; B-operand from Vl[d][t])
#pragma unroll
        for (int ks = 0; ks < 2; ++ks) {
            v16bf ap = load_a_frag<QSTR>(Qatt, wave * 16 + ln15, ks * 32, lane);
#pragma unroll
            for (int nt = 0; nt < 4; ++nt) {
                v16bf bv = load_b_frag<QSTR>(Vl, nt * 16 + ln15, ks * 32, lane);
                oacc[nt] = wmma_bf16(ap, bv, oacc[nt]);
            }
        }
    }

    // write O as bf16 in [B,S,D] (heads re-concatenated) for the final GEMM
    const int b = bh >> 5, h = bh & 31;
#pragma unroll
    for (int nt = 0; nt < 4; ++nt)
#pragma unroll
        for (int r = 0; r < 8; ++r) {
            const int s = sBase + wave * 16 + r + half * 8;
            const int d = nt * 16 + ln15;
            O[((size_t)b * 2048 + s) * 2048 + h * 64 + d] = f2bf(oacc[nt][r]);
        }
}

// ---------------------------------------------------------------------------
// host: orchestrate the pipeline on `stream`
// ---------------------------------------------------------------------------
extern "C" void kernel_launch(void* const* d_in, const int* in_sizes, int n_in,
                              void* d_out, int out_size, void* d_ws, size_t ws_size,
                              hipStream_t stream) {
    (void)in_sizes; (void)n_in; (void)out_size; (void)ws_size;
    const float* hs = (const float*)d_in[0];
    // d_in[1] = attention_mask (multiplicative tril) — implemented analytically
    const float* wq = (const float*)d_in[2];
    const float* bq = (const float*)d_in[3];
    const float* wk = (const float*)d_in[4];
    const float* bk = (const float*)d_in[5];
    const float* wv = (const float*)d_in[6];
    const float* bv = (const float*)d_in[7];
    const float* wo = (const float*)d_in[8];
    const float* bo = (const float*)d_in[9];

    const int Bc = 2, S = 2048, D = 2048, Hh = 32;
    const size_t MS = (size_t)Bc * S;            // 4096

    unsigned short* Xb  = (unsigned short*)d_ws;          // 4096*2048
    unsigned short* Wqb = Xb  + MS * D;
    unsigned short* Wkb = Wqb + (size_t)D * D;
    unsigned short* Wvb = Wkb + (size_t)D * D;
    unsigned short* Wob = Wvb + (size_t)D * D;
    unsigned short* Qh  = Wob + (size_t)D * D;            // [B,H,S,HD] bf16
    unsigned short* Kh  = Qh  + MS * D;
    unsigned short* Vh  = Kh  + MS * D;
    unsigned short* Ob  = Vh  + MS * D;                   // [B,S,D] bf16

    f32_to_bf16<<<4096, 256, 0, stream>>>(hs, Xb,  (int)(MS * D));
    f32_to_bf16<<<2048, 256, 0, stream>>>(wq, Wqb, D * D);
    f32_to_bf16<<<2048, 256, 0, stream>>>(wk, Wkb, D * D);
    f32_to_bf16<<<2048, 256, 0, stream>>>(wv, Wvb, D * D);
    f32_to_bf16<<<2048, 256, 0, stream>>>(wo, Wob, D * D);

    dim3 gGemm(D / 64, (unsigned)(MS / 128));             // (32, 32)
    gemm_bf16_ep<0><<<gGemm, 256, 0, stream>>>(Xb, Wqb, bq, Qh, (int)MS, D, D);
    gemm_bf16_ep<1><<<gGemm, 256, 0, stream>>>(Xb, Wkb, bk, Kh, (int)MS, D, D);
    gemm_bf16_ep<2><<<gGemm, 256, 0, stream>>>(Xb, Wvb, bv, Vh, (int)MS, D, D);

    dim3 gAttn(S / 128, Bc * Hh);                         // (16, 64)
    la_attention<<<gAttn, 256, 0, stream>>>(Qh, Kh, Vh, Ob);

    gemm_bf16_ep<3><<<gGemm, 256, 0, stream>>>(Ob, Wob, bo, d_out, (int)MS, D, D);
}
